// Q_ResNet18_5909874999650
// MI455X (gfx1250) — 8-bit matrix operand order repaired and checked by reading, compiled, not run on hardware
//
#include <hip/hip_runtime.h>


typedef __attribute__((ext_vector_type(8))) int v8i;
typedef __attribute__((ext_vector_type(4))) int v4i;

#define SA_SCALE (8.0f / 127.0f)
#define SA_INV   (127.0f / 8.0f)

enum {
  MODE_F32        = 0,
  MODE_RELU_Q8    = 1,
  MODE_ADD_I8_Q8  = 2,
  MODE_ADD_F32_Q8 = 3,
  MODE_Q8_SIGNED  = 4
};

__global__ void k_quant_input(const float* __restrict__ x, int* __restrict__ out,
                              int NHW, int HW)
{
  int i = blockIdx.x * blockDim.x + threadIdx.x;
  if (i >= NHW) return;
  int n = i / HW, hw = i - n * HW;
  const float* xp = x + (size_t)n * 3 * HW + hw;
  unsigned int pack = 0;
#pragma unroll
  for (int c = 0; c < 3; ++c) {
    float q = rintf(xp[(size_t)c * HW] * SA_INV);
    q = fminf(fmaxf(q, -128.0f), 127.0f);
    pack |= ((unsigned int)((int)q & 255)) << (8 * c);
  }
  out[i] = (int)pack;
}

__global__ void k_wscale(const float* __restrict__ w,
                         const float* __restrict__ gamma,
                         const float* __restrict__ beta,
                         const float* __restrict__ mean,
                         const float* __restrict__ var,
                         const float* __restrict__ fcb,
                         int OC, int K,
                         float* __restrict__ ws_out,
                         float* __restrict__ g_out,
                         float* __restrict__ b_out)
{
  __shared__ float red[256];
  int oc = blockIdx.x;
  float gg = 0.0f, bb = 0.0f;
  if (oc < OC) {
    if (gamma) {
      float s = sqrtf(var[oc] + 1e-5f);
      gg = gamma[oc] / s;
      bb = beta[oc] - mean[oc] * gg;
    } else {
      gg = 1.0f;
      bb = fcb ? fcb[oc] : 0.0f;
    }
  }
  float m = 0.0f;
  if (oc < OC) {
    const float* wp = w + (size_t)oc * K;
    for (int k = threadIdx.x; k < K; k += blockDim.x) m = fmaxf(m, fabsf(wp[k]));
  }
  red[threadIdx.x] = m;
  __syncthreads();
  for (int s = 128; s > 0; s >>= 1) {
    if ((int)threadIdx.x < s) red[threadIdx.x] = fmaxf(red[threadIdx.x], red[threadIdx.x + s]);
    __syncthreads();
  }
  if (threadIdx.x == 0) {
    float wsv = (oc < OC) ? red[0] * fabsf(gg) / 127.0f + 1e-8f : 1.0f;
    ws_out[blockIdx.x] = wsv;
    g_out[blockIdx.x]  = gg;
    b_out[blockIdx.x]  = bb;
  }
}

__global__ void k_wpack(const float* __restrict__ w,
                        const float* __restrict__ g,
                        const float* __restrict__ ws,
                        unsigned int* __restrict__ out,
                        int OC, int CIN, int CINp, int KH, int KW,
                        int nchunk, int ndwords)
{
  int d = blockIdx.x * blockDim.x + threadIdx.x;
  if (d >= ndwords) return;
  int byteBase = d * 4;
  int blk    = byteBase >> 10;
  int within = byteBase & 1023;
  int lane   = within >> 5;
  int v      = (within >> 2) & 7;
  int tile   = blk / nchunk;
  int chunk  = blk - tile * nchunk;
  int oc  = tile * 16 + (lane & 15);
  int hi  = lane >> 4;
  int Kst = chunk * 64 + ((v >> 2) << 5) + (hi << 4) + ((v & 3) << 2);
  int KHW = KH * KW;
  unsigned int pack = 0;
  if (oc < OC) {
    float invws = 1.0f / ws[oc];
    float gg = g[oc];
#pragma unroll
    for (int b = 0; b < 4; ++b) {
      int k = Kst + b;
      int s = k / CINp, ic = k - s * CINp;
      int q = 0;
      if (s < KHW && ic < CIN) {
        int kh = s / KW, kw = s - kh * KW;
        float wf = w[(((size_t)oc * CIN + ic) * KH + kh) * KW + kw] * gg;
        float qq = rintf(wf * invws);
        qq = fminf(fmaxf(qq, -128.0f), 127.0f);
        q = (int)qq;
      }
      pack |= ((unsigned int)(q & 255)) << (8 * b);
    }
  }
  out[d] = pack;
}

template <int MODE>
__global__ void __launch_bounds__(128) k_conv_wmma(
    const int* __restrict__ Apack, const signed char* __restrict__ act,
    const float* __restrict__ ws, const float* __restrict__ bias,
    int Nimg, int IH, int IW, int cinShift,
    int OH, int OW, int stride, int pad, int KW, int KHW,
    int nchunk, int octiles, int OC_valid, int OC_st,
    const void* __restrict__ ident, void* __restrict__ outp)
{
  __shared__ __align__(16) int ldsB[2][1024];
  __shared__ int ldsKT[64];

  const int tid  = threadIdx.x;
  const int lane = tid & 31;
  const int wv   = tid >> 5;
  const int CINp   = 1 << cinShift;
  const int icMask = CINp - 1;

  if (tid < KHW) {
    int kh = tid / KW, kw = tid - kh * KW;
    ldsKT[tid] = kh | (kw << 16);
  }

  const int octile  = blockIdx.y * 4 + wv;
  const int pixbase = blockIdx.x * 64;
  const int OHW = OH * OW;
  const int P   = Nimg * OHW;

  const int pcol = tid >> 1;
  const int kq0  = (tid & 1) << 3;
  const int fpix = pixbase + pcol;
  const bool fvalid = fpix < P;
  int ihb = 0, iwb = 0;
  int fimg = 0;
  if (fvalid) {
    fimg = fpix / OHW;
    int r  = fpix - fimg * OHW;
    int oh = r / OW, ow = r - oh * OW;
    ihb = oh * stride - pad;
    iwb = ow * stride - pad;
  }
  const signed char* frow = act + ((size_t)fimg * IH * IW << cinShift);

  const int octA = (octile < octiles) ? octile : 0;
  const int* Abase = Apack + (size_t)octA * nchunk * 256;

  auto fill = [&](int chunk, int buf) {
    int tmp[8];
#pragma unroll
    for (int i = 0; i < 8; ++i) {
      int kq = kq0 + i;
      int k  = (chunk << 6) + (kq << 2);
      int s  = k >> cinShift;
      int ic = k & icMask;
      int val = 0;
      if (fvalid && s < KHW) {
        int e  = ldsKT[s];
        int ih = ihb + (e & 0xffff);
        int iw = iwb + (e >> 16);
        if (ih >= 0 && ih < IH && iw >= 0 && iw < IW)
          val = *(const int*)(frow + ((((size_t)ih * IW + iw) << cinShift) + ic));
      }
      tmp[i] = val;
    }
    v4i* dst = (v4i*)&ldsB[buf][(pcol << 4) + kq0];
    dst[0] = (v4i){tmp[0], tmp[1], tmp[2], tmp[3]};
    dst[1] = (v4i){tmp[4], tmp[5], tmp[6], tmp[7]};
  };

  v8i acc[4];
#pragma unroll
  for (int st = 0; st < 4; ++st) acc[st] = (v8i){0, 0, 0, 0, 0, 0, 0, 0};

  const int ncol = lane & 15;
  const int hi   = lane >> 4;

  __syncthreads();
  fill(0, 0);

  for (int chunk = 0; chunk < nchunk; ++chunk) {
    __syncthreads();

    if (chunk + 1 < nchunk) fill(chunk + 1, (chunk + 1) & 1);

    v8i a = *(const v8i*)(Abase + (size_t)chunk * 256 + lane * 8);
    if (chunk + 1 < nchunk)
      __builtin_prefetch(Abase + (size_t)(chunk + 1) * 256 + lane * 8, 0, 3);

    const int* Bt = ldsB[chunk & 1];
#pragma unroll
    for (int st = 0; st < 4; ++st) {
      const v4i* bp = (const v4i*)&Bt[(((st << 4) + ncol) << 4) + (hi << 2)];
      v4i blo = bp[0];
      v4i bhi = bp[2];
      v8i b;
      b[0] = blo[0]; b[1] = blo[1]; b[2] = blo[2]; b[3] = blo[3];
      b[4] = bhi[0]; b[5] = bhi[1]; b[6] = bhi[2]; b[7] = bhi[3];
      acc[st] = __builtin_amdgcn_wmma_i32_16x16x64_iu8(true, a, true, b,
                                                       acc[st], false, false);
    }
  }

  if (octile >= octiles) return;

  float sv[8], bv[8];
#pragma unroll
  for (int r = 0; r < 8; ++r) {
    int oc = octile * 16 + hi * 8 + r;
    bool okc = oc < OC_valid;
    sv[r] = okc ? SA_SCALE * ws[oc] : 0.0f;
    bv[r] = okc ? bias[oc] : 0.0f;
  }

#pragma unroll
  for (int st = 0; st < 4; ++st) {
    int pix = pixbase + st * 16 + ncol;
    if (pix >= P) continue;
#pragma unroll
    for (int r = 0; r < 8; ++r) {
      int oc = octile * 16 + hi * 8 + r;
      if (oc >= OC_valid) continue;
      float y = (float)acc[st][r] * sv[r] + bv[r];
      size_t oidx = (size_t)pix * OC_st + oc;
      if (MODE == MODE_F32) {
        ((float*)outp)[oidx] = y;
      } else if (MODE == MODE_Q8_SIGNED) {
        float q = rintf(y * SA_INV);
        q = fminf(fmaxf(q, -128.0f), 127.0f);
        ((signed char*)outp)[oidx] = (signed char)(int)q;
      } else {
        if (MODE == MODE_ADD_I8_Q8)
          y += (float)((const signed char*)ident)[oidx] * SA_SCALE;
        else if (MODE == MODE_ADD_F32_Q8)
          y += ((const float*)ident)[oidx];
        float q = rintf(y * SA_INV);
        q = fminf(fmaxf(q, 0.0f), 127.0f);
        ((signed char*)outp)[oidx] = (signed char)(int)q;
      }
    }
  }
}

__global__ void k_maxpool_q8(const signed char* __restrict__ in,
                             signed char* __restrict__ out,
                             int Nimg, int IH, int IW, int C, int OH, int OW)
{
  int C4 = C >> 2;
  int total = Nimg * OH * OW * C4;
  int i = blockIdx.x * blockDim.x + threadIdx.x;
  if (i >= total) return;
  int c4 = i % C4;
  int r  = i / C4;
  int ow = r % OW; r /= OW;
  int oh = r % OH;
  int n  = r / OH;
  int m0 = 0, m1 = 0, m2 = 0, m3 = 0;
  for (int kh = 0; kh < 3; ++kh) {
    int ih = oh * 2 - 1 + kh;
    if (ih < 0 || ih >= IH) continue;
    for (int kw = 0; kw < 3; ++kw) {
      int iw = ow * 2 - 1 + kw;
      if (iw < 0 || iw >= IW) continue;
      int dv = *(const int*)(in + (((size_t)n * IH + ih) * IW + iw) * C + c4 * 4);
      m0 = max(m0, (int)(signed char)(dv & 255));
      m1 = max(m1, (int)(signed char)((dv >> 8) & 255));
      m2 = max(m2, (int)(signed char)((dv >> 16) & 255));
      m3 = max(m3, (int)(signed char)((dv >> 24) & 255));
    }
  }
  unsigned int pack = (unsigned)(m0 & 255) | ((unsigned)(m1 & 255) << 8) |
                      ((unsigned)(m2 & 255) << 16) | ((unsigned)(m3 & 255) << 24);
  *(int*)(out + (((size_t)n * OH + oh) * OW + ow) * C + c4 * 4) = (int)pack;
}

__global__ void k_avgpool_q8(const signed char* __restrict__ in,
                             signed char* __restrict__ out,
                             int Nimg, int HW, int C)
{
  int i = blockIdx.x * blockDim.x + threadIdx.x;
  if (i >= Nimg * C) return;
  int c = i % C, n = i / C;
  int sum = 0;
  for (int s = 0; s < HW; ++s) sum += (int)in[((size_t)n * HW + s) * C + c];
  float q = rintf((float)sum / (float)HW);
  q = fminf(fmaxf(q, -128.0f), 127.0f);
  out[i] = (signed char)(int)q;
}

struct RawConv { const float *beta, *gamma, *mean, *var, *w; };

struct ConvPrep {
  float* ws; float* g; float* bias;
  int* apack;
  int OC, CIN, CINp, cinShift, KH, KW, nchunk, octiles;
};

extern "C" void kernel_launch(void* const* d_in, const int* in_sizes, int n_in,
                              void* d_out, int out_size, void* d_ws, size_t ws_size,
                              hipStream_t stream)
{
  (void)in_sizes; (void)n_in; (void)out_size; (void)ws_size;

  int pi = 0;
  pi++;
  const bool hasid[8] = {false, false, true, false, true, false, true, false};
  const int  chs[8]   = {64, 64, 128, 128, 256, 256, 512, 512};
  const int  strd[8]  = {1, 1, 2, 1, 2, 1, 2, 1};
  RawConv bconv[8][3];
  for (int i = 0; i < 8; ++i) {
    int nc = hasid[i] ? 3 : 2;
    for (int j = 0; j < nc; ++j) {
      bconv[i][j].beta  = (const float*)d_in[pi++];
      bconv[i][j].gamma = (const float*)d_in[pi++];
      bconv[i][j].mean  = (const float*)d_in[pi++];
      bconv[i][j].var   = (const float*)d_in[pi++];
      bconv[i][j].w     = (const float*)d_in[pi++];
    }
  }
  const float* fc_b = (const float*)d_in[pi++];
  const float* fc_w = (const float*)d_in[pi++];
  RawConv initc;
  initc.beta  = (const float*)d_in[pi++];
  initc.gamma = (const float*)d_in[pi++];
  initc.mean  = (const float*)d_in[pi++];
  initc.var   = (const float*)d_in[pi++];
  initc.w     = (const float*)d_in[pi++];
  const float* x = (const float*)d_in[pi++];

  char* wsbase = (char*)d_ws;
  size_t off = 0;
  auto walloc = [&](size_t bytes) -> void* {
    off = (off + 255) & ~(size_t)255;
    void* p = wsbase + off;
    off += bytes;
    return p;
  };

  signed char* xq8   = (signed char*)walloc((size_t)32 * 224 * 224 * 4);
  signed char* c1out = (signed char*)walloc((size_t)32 * 112 * 112 * 64);
  signed char* actA  = (signed char*)walloc((size_t)32 * 56 * 56 * 64);
  signed char* actB  = (signed char*)walloc((size_t)32 * 56 * 56 * 64);
  signed char* actT  = (signed char*)walloc((size_t)32 * 56 * 56 * 64);
  float*       idf32 = (float*)walloc((size_t)32 * 28 * 28 * 128 * 4);
  signed char* fcin8 = (signed char*)walloc((size_t)32 * 512);

  auto make_prep = [&](const float* w, const float* gamma, const float* beta,
                       const float* mean, const float* var, const float* fcb,
                       int OC, int CIN, int KH, int KW) -> ConvPrep {
    ConvPrep cp;
    cp.OC = OC; cp.CIN = CIN; cp.CINp = (CIN == 3) ? 4 : CIN;
    cp.cinShift = __builtin_ctz((unsigned)cp.CINp);
    cp.KH = KH; cp.KW = KW;
    int Ktot = KH * KW * cp.CINp;
    cp.nchunk  = (Ktot + 63) / 64;
    cp.octiles = (OC + 15) / 16;
    int OCpad = cp.octiles * 16;
    cp.ws    = (float*)walloc((size_t)OCpad * 4);
    cp.g     = (float*)walloc((size_t)OCpad * 4);
    cp.bias  = (float*)walloc((size_t)OCpad * 4);
    cp.apack = (int*)walloc((size_t)cp.octiles * cp.nchunk * 1024);
    k_wscale<<<OCpad, 256, 0, stream>>>(w, gamma, beta, mean, var, fcb,
                                        OC, CIN * KH * KW, cp.ws, cp.g, cp.bias);
    int ndw = cp.octiles * cp.nchunk * 256;
    k_wpack<<<(ndw + 255) / 256, 256, 0, stream>>>(
        w, cp.g, cp.ws, (unsigned int*)cp.apack,
        OC, CIN, cp.CINp, KH, KW, cp.nchunk, ndw);
    return cp;
  };

  ConvPrep p_init = make_prep(initc.w, initc.gamma, initc.beta, initc.mean,
                              initc.var, nullptr, 64, 3, 7, 7);
  ConvPrep p_blk[8][3];
  int cin = 64;
  for (int i = 0; i < 8; ++i) {
    p_blk[i][0] = make_prep(bconv[i][0].w, bconv[i][0].gamma, bconv[i][0].beta,
                            bconv[i][0].mean, bconv[i][0].var, nullptr,
                            chs[i], cin, 3, 3);
    p_blk[i][1] = make_prep(bconv[i][1].w, bconv[i][1].gamma, bconv[i][1].beta,
                            bconv[i][1].mean, bconv[i][1].var, nullptr,
                            chs[i], chs[i], 3, 3);
    if (hasid[i])
      p_blk[i][2] = make_prep(bconv[i][2].w, bconv[i][2].gamma, bconv[i][2].beta,
                              bconv[i][2].mean, bconv[i][2].var, nullptr,
                              chs[i], cin, 1, 1);
    cin = chs[i];
  }
  ConvPrep p_fc = make_prep(fc_w, nullptr, nullptr, nullptr, nullptr, fc_b,
                            1000, 512, 1, 1);

  auto run_conv = [&](const ConvPrep& cp, const signed char* act,
                      int Nimg, int IH, int IW, int OH, int OW,
                      int stride, int pad, int mode,
                      const void* ident, void* outp, int OC_valid, int OC_st) {
    int P = Nimg * OH * OW;
    dim3 grid((P + 63) / 64, (cp.octiles + 3) / 4);
#define CONV_ARGS cp.apack, act, cp.ws, cp.bias, Nimg, IH, IW, cp.cinShift, \
                  OH, OW, stride, pad, cp.KW, cp.KH * cp.KW, cp.nchunk,     \
                  cp.octiles, OC_valid, OC_st, ident, outp
    switch (mode) {
      case MODE_F32:        k_conv_wmma<MODE_F32><<<grid, 128, 0, stream>>>(CONV_ARGS);        break;
      case MODE_RELU_Q8:    k_conv_wmma<MODE_RELU_Q8><<<grid, 128, 0, stream>>>(CONV_ARGS);    break;
      case MODE_ADD_I8_Q8:  k_conv_wmma<MODE_ADD_I8_Q8><<<grid, 128, 0, stream>>>(CONV_ARGS);  break;
      case MODE_ADD_F32_Q8: k_conv_wmma<MODE_ADD_F32_Q8><<<grid, 128, 0, stream>>>(CONV_ARGS); break;
      default:              k_conv_wmma<MODE_Q8_SIGNED><<<grid, 128, 0, stream>>>(CONV_ARGS);  break;
    }
#undef CONV_ARGS
  };

  {
    int NHW = 32 * 224 * 224;
    k_quant_input<<<(NHW + 255) / 256, 256, 0, stream>>>(x, (int*)xq8, NHW, 224 * 224);
  }
  run_conv(p_init, xq8, 32, 224, 224, 112, 112, 2, 3, MODE_Q8_SIGNED,
           nullptr, c1out, 64, 64);
  {
    int total = 32 * 56 * 56 * (64 / 4);
    k_maxpool_q8<<<(total + 255) / 256, 256, 0, stream>>>(c1out, actA,
                                                          32, 112, 112, 64, 56, 56);
  }

  int H = 56;
  for (int i = 0; i < 8; ++i) {
    int OC = chs[i], s = strd[i];
    int OHo = (H + 2 - 3) / s + 1;
    if (hasid[i])
      run_conv(p_blk[i][2], actA, 32, H, H, OHo, OHo, s, 0, MODE_F32,
               nullptr, idf32, OC, OC);
    run_conv(p_blk[i][0], actA, 32, H, H, OHo, OHo, s, 1, MODE_RELU_Q8,
             nullptr, actT, OC, OC);
    if (hasid[i])
      run_conv(p_blk[i][1], actT, 32, OHo, OHo, OHo, OHo, 1, 1, MODE_ADD_F32_Q8,
               idf32, actB, OC, OC);
    else
      run_conv(p_blk[i][1], actT, 32, OHo, OHo, OHo, OHo, 1, 1, MODE_ADD_I8_Q8,
               actA, actB, OC, OC);
    signed char* t = actA; actA = actB; actB = t;
    H = OHo;
  }

  k_avgpool_q8<<<(32 * 512 + 255) / 256, 256, 0, stream>>>(actA, fcin8, 32, 49, 512);

  run_conv(p_fc, fcin8, 32, 1, 1, 1, 1, 1, 0, MODE_F32,
           nullptr, (float*)d_out, 1000, 1000);
}
